// EncoderBlock_74285754351786
// MI455X (gfx1250) — matrix operand order repaired and checked by reading, not compiled, not run on hardware
//
#include <hip/hip_runtime.h>
#include <hip/hip_bf16.h>
#include <cstdint>
#include <cstddef>

#define B_  2
#define S_  2048
#define D_  1024
#define H_  16
#define F_  4096
#define HD_ 64
#define M_  (B_ * S_)          // 4096 rows of tokens

typedef __attribute__((ext_vector_type(16))) __bf16          bf16x16;
typedef __attribute__((ext_vector_type(8)))  float           f32x8;
typedef __attribute__((ext_vector_type(16))) unsigned short  u16x16;

union Frag {
    bf16x16 bf;
    u16x16  us;
    uint4   q[2];
};

__device__ inline unsigned short f2bf(float f) {
    unsigned int u = __builtin_bit_cast(unsigned int, f);
    unsigned int r = u + 0x7FFFu + ((u >> 16) & 1u);   // round-to-nearest-even
    return (unsigned short)(r >> 16);
}

__device__ inline f32x8 wmma_bf16(bf16x16 a, bf16x16 b, f32x8 c) {
    // (neg_a, A, neg_b, B, c_mod, C, reuse_a, reuse_b)
    return __builtin_amdgcn_wmma_f32_16x16x32_bf16(false, a, false, b, (short)0, c, false, false);
}

// Async DMA: global memory -> LDS, 16 bytes per lane, tracked by ASYNCcnt.
// vdst = VGPR holding LDS byte offset; addr = per-lane 64-bit global address.
__device__ inline void async_g2lds_b128(unsigned int lds_off, const void* gaddr) {
    asm volatile("global_load_async_to_lds_b128 %0, %1, off"
                 :: "v"(lds_off), "v"(gaddr) : "memory");
}
__device__ inline void wait_asynccnt_1() { asm volatile("s_wait_asynccnt 0x1" ::: "memory"); }
__device__ inline void wait_asynccnt_0() { asm volatile("s_wait_asynccnt 0x0" ::: "memory"); }

__device__ inline unsigned int lds_off32(const void* p) {
    return (unsigned int)(unsigned long long)p;   // flat->LDS aperture keeps offset in [31:0]
}

// ---------------------------------------------------------------------------
// fp32 -> bf16 conversion
// ---------------------------------------------------------------------------
__global__ __launch_bounds__(256) void k_f32_to_bf16(const float* __restrict__ in,
                                                     unsigned short* __restrict__ out,
                                                     int n) {
    int i = blockIdx.x * 256 + threadIdx.x;
    if (i < n) out[i] = f2bf(in[i]);
}

// ---------------------------------------------------------------------------
// Pack V columns: vt[b][h][d][s] = qkv[b*S+s][h*3*HD + 2*HD + d]
// ---------------------------------------------------------------------------
__global__ __launch_bounds__(256) void k_repack_vt(const unsigned short* __restrict__ qkv,
                                                   unsigned short* __restrict__ vt) {
    int i = blockIdx.x * 256 + threadIdx.x;          // B*H*S*HD = 4M
    if (i >= B_ * H_ * S_ * HD_) return;
    int d = i & (HD_ - 1);
    int s = (i >> 6) & (S_ - 1);
    int h = (i >> 17) & (H_ - 1);
    int b = i >> 21;
    vt[((size_t)((b * H_ + h) * HD_ + d)) * S_ + s] =
        qkv[(size_t)(b * S_ + s) * (3 * D_) + h * (3 * HD_) + 2 * HD_ + d];
}

// ---------------------------------------------------------------------------
// bf16 GEMM: C[M,N] = act(A[M,K] @ W[N,K]^T + bias [+ res])
// flags: 1=ReLU, 2=write bf16 (Cb), 4=add residual (res[M,N] fp32)
// block: 256 threads = 8 waves; block tile 256(M) x 64(N); wave tile 32 x 64.
// W tiles double-buffered in LDS via async global->LDS DMA (ASYNCcnt); the
// DMA of K-step k+32 overlaps the 8 WMMAs of step k. Each B fragment feeds
// two WMMAs (two A row sub-tiles) -> 8 WMMAs per 4 LDS fragment reads.
// Last K-step is peeled so the hot loop has straight-line control flow.
// ---------------------------------------------------------------------------
__global__ __launch_bounds__(256)
void k_gemm_bf16(const unsigned short* __restrict__ A,
                 const unsigned short* __restrict__ W,
                 const float* __restrict__ bias,
                 const float* __restrict__ res,
                 float* __restrict__ Cf,
                 unsigned short* __restrict__ Cb,
                 int M, int N, int K, int flags) {
    __shared__ __align__(32) unsigned short Bs[2][64 * 32];   // 2 x 4 KB W tiles

    const int tid     = threadIdx.x;
    const int wid     = tid >> 5;
    const int lane    = tid & 31;
    const int lhalf   = lane >> 4;     // 0|1
    const int l16     = lane & 15;
    const int rowbase = blockIdx.y * 256 + wid * 32;
    const int colbase = blockIdx.x * 64;

    const int r0 = tid >> 2;           // staging row (0..63)
    const int c0 = (tid & 3) * 8;      // staging k offset (halfs)

    const unsigned short* wstage = &W[(size_t)(colbase + r0) * K + c0];
    const unsigned int stage_off[2] = { lds_off32(&Bs[0][r0 * 32 + c0]),
                                        lds_off32(&Bs[1][r0 * 32 + c0]) };

    f32x8 acc[2][4] = {};

    auto step = [&](int kk, int cur) {
        // A fragments, two 16-row sub-tiles; lane half selects K sub-chunks
        const unsigned short* ap0 = &A[(size_t)(rowbase + l16) * K + kk + lhalf * 8];
        const unsigned short* ap1 = ap0 + (size_t)16 * K;
        Frag a0, a1;
        a0.q[0] = *(const uint4*)(ap0);
        a0.q[1] = *(const uint4*)(ap0 + 16);
        a1.q[0] = *(const uint4*)(ap1);
        a1.q[1] = *(const uint4*)(ap1 + 16);
        __builtin_prefetch(ap0 + 32, 0, 1);   // speculative; OOB is dropped
        __builtin_prefetch(ap1 + 32, 0, 1);
#pragma unroll
        for (int j = 0; j < 4; ++j) {
            const unsigned short* bp = &Bs[cur][(j * 16 + l16) * 32 + lhalf * 8];
            Frag b;
            b.q[0] = ((const uint4*)bp)[0];
            b.q[1] = ((const uint4*)bp)[2];
            acc[0][j] = wmma_bf16(a0.bf, b.bf, acc[0][j]);
            acc[1][j] = wmma_bf16(a1.bf, b.bf, acc[1][j]);
        }
    };

    // prologue: kick off DMA of the first W tile
    async_g2lds_b128(stage_off[0], wstage);

    int kk = 0;
    for (; kk < K - 32; kk += 32) {          // straight-line hot loop
        const int cur = (kk >> 5) & 1;
        async_g2lds_b128(stage_off[cur ^ 1], wstage + kk + 32);
        wait_asynccnt_1();                   // current tile landed (in-order completion)
        __syncthreads();
        step(kk, cur);
        __syncthreads();                     // readers done before DMA overwrites Bs[cur]
    }
    {                                        // peeled last K-step
        const int cur = (kk >> 5) & 1;
        wait_asynccnt_0();
        __syncthreads();
        step(kk, cur);
    }

    // epilogue: lane holds column n=l16 (+16j), rows m = rowbase + i*16 + lhalf*8 + r
#pragma unroll
    for (int i = 0; i < 2; ++i) {
#pragma unroll
        for (int j = 0; j < 4; ++j) {
            const int n  = colbase + j * 16 + l16;
            const float bv = bias[n];
#pragma unroll
            for (int r = 0; r < 8; ++r) {
                const int m = rowbase + i * 16 + lhalf * 8 + r;
                float v = acc[i][j][r] + bv;
                if (flags & 4) v += res[(size_t)m * N + n];
                if (flags & 1) v = fmaxf(v, 0.f);
                if (flags & 2) Cb[(size_t)m * N + n] = f2bf(v);
                else           Cf[(size_t)m * N + n] = v;
            }
        }
    }
}

// ---------------------------------------------------------------------------
// Flash attention: one wave per (b, h, 16-row q tile).
// S^T = K @ Q^T so C-layout of S^T == A-layout of P for O += P @ V.
// ---------------------------------------------------------------------------
__global__ __launch_bounds__(256)
void k_attn(const unsigned short* __restrict__ qkv,   // [B*S, 3D] bf16
            const unsigned short* __restrict__ vt,    // [B,H,HD,S] bf16
            unsigned short* __restrict__ ctx) {       // [B*S, D] bf16
    const int wid   = threadIdx.x >> 5;
    const int lane  = threadIdx.x & 31;
    const int gw    = blockIdx.x * 8 + wid;           // 4096 q-tiles total
    const int qt    = gw & 127;                       // S/16
    const int h     = (gw >> 7) & 15;
    const int b     = gw >> 11;
    const int lhalf = lane >> 4;
    const int l16   = lane & 15;
    const int qrow  = qt * 16;

    // Q as B-operand: lane = q column, d = 8*lhalf+(0..7) and 16+8*lhalf+(0..7) per 32-d step
    Frag qf[2];
    {
        const unsigned short* qp =
            qkv + (size_t)(b * S_ + qrow + l16) * (3 * D_) + h * (3 * HD_);
#pragma unroll
        for (int ds = 0; ds < 2; ++ds) {
            const unsigned short* p = qp + ds * 32 + lhalf * 8;
            qf[ds].q[0] = ((const uint4*)p)[0];
            qf[ds].q[1] = ((const uint4*)p)[2];
        }
    }

    f32x8 o[4] = {};
    float mrun = -1e30f, lrun = 0.f;

    for (int kc = 0; kc < S_; kc += 32) {
        f32x8 st[2] = {};
        // S^T tiles: rows = k indices (two 16-row tiles), cols = q
#pragma unroll
        for (int ds = 0; ds < 2; ++ds) {
#pragma unroll
            for (int t = 0; t < 2; ++t) {
                const unsigned short* kp =
                    qkv + (size_t)(b * S_ + kc + t * 16 + l16) * (3 * D_) +
                    h * (3 * HD_) + HD_ + ds * 32 + lhalf * 8;
                Frag ka;
                ka.q[0] = *(const uint4*)(kp);
                ka.q[1] = *(const uint4*)(kp + 16);
                st[t] = wmma_bf16(ka.bf, qf[ds].bf, st[t]);
            }
        }

        // online softmax; lane's q = l16, its k rows = lhalf*8 + r (+16 for st[1])
        float s0[8], s1[8];
        float mx = -1e30f;
#pragma unroll
        for (int r = 0; r < 8; ++r) {
            s0[r] = st[0][r] * 0.125f;     // 1/sqrt(64)
            s1[r] = st[1][r] * 0.125f;
            mx = fmaxf(mx, fmaxf(s0[r], s1[r]));
        }
        mx = fmaxf(mx, __shfl_xor(mx, 16));
        const float mnew  = fmaxf(mrun, mx);
        const float alpha = __expf(mrun - mnew);

        Frag pf;
        float psum = 0.f;
#pragma unroll
        for (int r = 0; r < 8; ++r) {
            const float e0 = __expf(s0[r] - mnew);
            const float e1 = __expf(s1[r] - mnew);
            psum += e0 + e1;
            pf.us[r]     = f2bf(e0);       // A-layout K = lhalf*8 + r
            pf.us[r + 8] = f2bf(e1);       // A-layout K = 16 + lhalf*8 + r
        }
        psum += __shfl_xor(psum, 16);
        lrun = lrun * alpha + psum;
        mrun = mnew;

        // rescale O (O rows q = lhalf*8 + r; stats live at lane q)
        float ar[8];
#pragma unroll
        for (int r = 0; r < 8; ++r) ar[r] = __shfl(alpha, lhalf * 8 + r);
#pragma unroll
        for (int j = 0; j < 4; ++j)
#pragma unroll
            for (int r = 0; r < 8; ++r) o[j][r] *= ar[r];

        // O += P @ V ; V^T rows give contiguous k for each d column
#pragma unroll
        for (int j = 0; j < 4; ++j) {
            const unsigned short* vp =
                vt + (size_t)((b * H_ + h) * HD_ + j * 16 + l16) * S_ +
                kc + lhalf * 8;
            Frag vf;
            vf.q[0] = ((const uint4*)vp)[0];
            vf.q[1] = ((const uint4*)vp)[2];
            o[j] = wmma_bf16(pf.bf, vf.bf, o[j]);
        }
    }

    const float inv = 1.f / lrun;
    float ir[8];
#pragma unroll
    for (int r = 0; r < 8; ++r) ir[r] = __shfl(inv, lhalf * 8 + r);
#pragma unroll
    for (int j = 0; j < 4; ++j)
#pragma unroll
        for (int r = 0; r < 8; ++r) {
            const size_t row = (size_t)(b * S_ + qrow + lhalf * 8 + r);
            ctx[row * D_ + h * HD_ + j * 16 + l16] = f2bf(o[j][r] * ir[r]);
        }
}

// ---------------------------------------------------------------------------
// LayerNorm over D=1024; one block per row; optional bf16 mirror output
// ---------------------------------------------------------------------------
__global__ __launch_bounds__(256)
void k_layernorm(const float* __restrict__ X, const float* __restrict__ g,
                 const float* __restrict__ bta, float* __restrict__ outf,
                 unsigned short* __restrict__ outb) {
    __shared__ float s1[256];
    __shared__ float s2[256];
    const int row = blockIdx.x;
    const int tid = threadIdx.x;
    const float4 x = ((const float4*)(X + (size_t)row * D_))[tid];
    s1[tid] = x.x + x.y + x.z + x.w;
    s2[tid] = x.x * x.x + x.y * x.y + x.z * x.z + x.w * x.w;
    __syncthreads();
    for (int off = 128; off > 0; off >>= 1) {
        if (tid < off) { s1[tid] += s1[tid + off]; s2[tid] += s2[tid + off]; }
        __syncthreads();
    }
    const float mean = s1[0] * (1.f / D_);
    const float var  = s2[0] * (1.f / D_) - mean * mean;
    const float rstd = rsqrtf(var + 1e-5f);
    const float4 gg = ((const float4*)g)[tid];
    const float4 bb = ((const float4*)bta)[tid];
    float4 y;
    y.x = (x.x - mean) * rstd * gg.x + bb.x;
    y.y = (x.y - mean) * rstd * gg.y + bb.y;
    y.z = (x.z - mean) * rstd * gg.z + bb.z;
    y.w = (x.w - mean) * rstd * gg.w + bb.w;
    ((float4*)(outf + (size_t)row * D_))[tid] = y;
    if (outb) {
        unsigned short* ob = outb + (size_t)row * D_ + tid * 4;
        ob[0] = f2bf(y.x); ob[1] = f2bf(y.y); ob[2] = f2bf(y.z); ob[3] = f2bf(y.w);
    }
}

// ---------------------------------------------------------------------------
extern "C" void kernel_launch(void* const* d_in, const int* in_sizes, int n_in,
                              void* d_out, int out_size, void* d_ws, size_t ws_size,
                              hipStream_t stream) {
    (void)in_sizes; (void)n_in; (void)out_size; (void)ws_size;
    const float* src    = (const float*)d_in[0];
    const float* qkv_w  = (const float*)d_in[1];
    const float* qkv_b  = (const float*)d_in[2];
    const float* out_w  = (const float*)d_in[3];
    const float* out_b  = (const float*)d_in[4];
    const float* ffn_w1 = (const float*)d_in[5];
    const float* ffn_b1 = (const float*)d_in[6];
    const float* ffn_w2 = (const float*)d_in[7];
    const float* ffn_b2 = (const float*)d_in[8];
    const float* ln1_g  = (const float*)d_in[9];
    const float* ln1_b  = (const float*)d_in[10];
    const float* ln2_g  = (const float*)d_in[11];
    const float* ln2_b  = (const float*)d_in[12];

    char* ws = (char*)d_ws;
    const size_t MB = 1ull << 20;
    unsigned short* srcb  = (unsigned short*)(ws + 0 * MB);     // 8 MB
    unsigned short* wqkvb = (unsigned short*)(ws + 8 * MB);     // 6 MB
    unsigned short* woutb = (unsigned short*)(ws + 14 * MB);    // 2 MB
    unsigned short* w1b   = (unsigned short*)(ws + 16 * MB);    // 8 MB
    unsigned short* w2b   = (unsigned short*)(ws + 24 * MB);    // 8 MB
    unsigned short* qkvb  = (unsigned short*)(ws + 32 * MB);    // 24 MB
    unsigned short* vt    = (unsigned short*)(ws + 56 * MB);    // 8 MB
    unsigned short* ctxb  = (unsigned short*)(ws + 64 * MB);    // 8 MB
    float*          X     = (float*)(ws + 72 * MB);             // 16 MB
    float*          agg   = (float*)(ws + 88 * MB);             // 16 MB
    unsigned short* aggb  = (unsigned short*)(ws + 104 * MB);   // 8 MB
    unsigned short* hb    = (unsigned short*)(ws + 32 * MB);    // 32 MB (reuses qkvb+vt)
    float*          Y     = (float*)(ws + 72 * MB);             // 16 MB (reuses X)

    // ---- fp32 -> bf16 conversions
    auto cvt = [&](const float* in, unsigned short* out, int n) {
        k_f32_to_bf16<<<(n + 255) / 256, 256, 0, stream>>>(in, out, n);
    };
    cvt(src,    srcb,  M_ * D_);
    cvt(qkv_w,  wqkvb, 3 * D_ * D_);
    cvt(out_w,  woutb, D_ * D_);
    cvt(ffn_w1, w1b,   F_ * D_);
    cvt(ffn_w2, w2b,   D_ * F_);

    // ---- QKV projection: qkv(bf16) = src @ qkv_w^T + qkv_b
    k_gemm_bf16<<<dim3(3 * D_ / 64, M_ / 256), 256, 0, stream>>>(
        srcb, wqkvb, qkv_b, nullptr, nullptr, qkvb, M_, 3 * D_, D_, /*flags=*/2);

    // ---- pack V transposed for contiguous B-operand loads
    k_repack_vt<<<(B_ * H_ * S_ * HD_) / 256, 256, 0, stream>>>(qkvb, vt);

    // ---- flash attention -> ctx (bf16)
    k_attn<<<(B_ * H_ * (S_ / 16)) / 8, 256, 0, stream>>>(qkvb, vt, ctxb);

    // ---- out projection + residual: X = ctx @ out_w^T + out_b + src
    k_gemm_bf16<<<dim3(D_ / 64, M_ / 256), 256, 0, stream>>>(
        ctxb, woutb, out_b, src, X, nullptr, M_, D_, D_, /*flags=*/4);

    // ---- LN1 -> agg (fp32) + aggb (bf16)
    k_layernorm<<<M_, 256, 0, stream>>>(X, ln1_g, ln1_b, agg, aggb);

    // ---- FFN1: h = relu(agg @ w1^T + b1) (bf16)
    k_gemm_bf16<<<dim3(F_ / 64, M_ / 256), 256, 0, stream>>>(
        aggb, w1b, ffn_b1, nullptr, nullptr, hb, M_, F_, D_, /*flags=*/1 | 2);

    // ---- FFN2 + residual: Y = h @ w2^T + b2 + agg
    k_gemm_bf16<<<dim3(D_ / 64, M_ / 256), 256, 0, stream>>>(
        hb, w2b, ffn_b2, agg, Y, nullptr, M_, D_, F_, /*flags=*/4);

    // ---- LN2 -> d_out (fp32)
    k_layernorm<<<M_, 256, 0, stream>>>(Y, ln2_g, ln2_b, (float*)d_out, nullptr);
}
